// ODEFunc_89300960019068
// MI455X (gfx1250) — matrix operand order repaired and checked by reading, not compiled, not run on hardware
//
#include <hip/hip_runtime.h>
#include <hip/hip_bf16.h>
#include <math.h>

typedef __attribute__((ext_vector_type(16))) __bf16 v16bf;
typedef __attribute__((ext_vector_type(8)))  float  v8f;

union BF16x16 { v16bf v; uint4 u[2]; };

#define M_TILE  128
#define HSTRIDE 264   // 256 + 8 bf16 pad: row pitch 528B = 33*16B -> conflict-free b128 LDS loads

__device__ __forceinline__ float  bf2f(__bf16 a){ return (float)a; }
__device__ __forceinline__ __bf16 f2bf(float a){ return (__bf16)a; }

// ---------------------------------------------------------------------------
// Kernel 0: convert weights to bf16 operand layouts.
//   W0T [256][32]  : B for layer1 fwd  (B[k,n]=W0[k,n], K padded 11->32)
//   W0P [16][256]  : B for final grad  (B[k=i,n=d]=W0[d,i], N padded 11->16)
//   W1T/W2T [256][256]: B for fwd layers (stored [n][k] = W[k][n])
//   W1B/W2B [256][256]: B for bwd layers (stored [n][k] = W[n][k] = row-major W)
// ---------------------------------------------------------------------------
__global__ __launch_bounds__(256) void prep_weights(
    const float* __restrict__ W0, const float* __restrict__ W1,
    const float* __restrict__ W2,
    __bf16* __restrict__ W0T, __bf16* __restrict__ W0P,
    __bf16* __restrict__ W1T, __bf16* __restrict__ W2T,
    __bf16* __restrict__ W1B, __bf16* __restrict__ W2B)
{
  const int idx = blockIdx.x * blockDim.x + threadIdx.x;   // 0..65535
  const int n = idx >> 8, k = idx & 255;
  W1T[idx] = f2bf(W1[k * 256 + n]);
  W2T[idx] = f2bf(W2[k * 256 + n]);
  W1B[idx] = f2bf(W1[idx]);
  W2B[idx] = f2bf(W2[idx]);
  if (idx < 256 * 32) {
    const int nn = idx >> 5, kk = idx & 31;
    W0T[idx] = (kk < 11) ? f2bf(W0[kk * 256 + nn]) : (__bf16)0.0f;
  }
  if (idx < 16 * 256) {
    const int d = idx >> 8, i = idx & 255;
    W0P[idx] = (d < 11) ? f2bf(W0[d * 256 + i]) : (__bf16)0.0f;
  }
}

// ---------------------------------------------------------------------------
// One GEMM phase: this wave's (MT*16 rows) x (NT*16 cols), K = KT*32.
// A in LDS (bf16, row-major, pitch astride), B in global (bf16, [N][K]).
// Each B tile feeds MT WMMAs, each A tile feeds NT WMMAs (operand reuse).
// EPI: 0 = tanh(z+bias) -> Hout(bf16)
//      1 = g3 = W3[n]*(1-tanh(z+b)^2) -> Hout
//      2 = g  = z*(1-h^2), h from Hmask; write over Hmask buffer
//      3 = raw f32 store to gout (grad_rq)
// ---------------------------------------------------------------------------
template<int KT, int MT, int NT, int EPI>
__device__ __forceinline__ void gemm_phase(
    const __bf16* __restrict__ A, int astride,
    const __bf16* __restrict__ Bg, int bstride,
    __bf16* __restrict__ Hout, const __bf16* __restrict__ Hmask,
    const float* __restrict__ bias, const float* __restrict__ w3,
    float* __restrict__ gout, int lane, int waveM, int waveN)
{
  const int half = lane >> 4;   // ISA 7.12.2: lane halves select K sub-ranges
  const int l15  = lane & 15;

  v8f acc[MT][NT];
  #pragma unroll
  for (int mt = 0; mt < MT; ++mt)
    #pragma unroll
    for (int nt = 0; nt < NT; ++nt)
      #pragma unroll
      for (int j = 0; j < 8; ++j) acc[mt][nt][j] = 0.0f;

  for (int kt = 0; kt < KT; ++kt) {
    // A 16x32 bf16 tiles: lanes 0-15 row M=lane K{0..7,16..23}; lanes 16-31 K{8..15,24..31}
    BF16x16 a[MT];
    #pragma unroll
    for (int mt = 0; mt < MT; ++mt) {
      const __bf16* ap = A + (size_t)(waveM + mt * 16 + l15) * astride + kt * 32 + half * 8;
      a[mt].u[0] = *(const uint4*)(ap);
      a[mt].u[1] = *(const uint4*)(ap + 16);
    }
    // B 32x16 bf16 tiles from [N][K] store: lane n -> K{0..7,16..23}, lane n+16 -> K{8..15,24..31}
    BF16x16 b[NT];
    #pragma unroll
    for (int nt = 0; nt < NT; ++nt) {
      const __bf16* bp = Bg + (size_t)(waveN + nt * 16 + l15) * bstride + kt * 32 + half * 8;
      b[nt].u[0] = *(const uint4*)(bp);
      b[nt].u[1] = *(const uint4*)(bp + 16);
    }
    #pragma unroll
    for (int nt = 0; nt < NT; ++nt)
      #pragma unroll
      for (int mt = 0; mt < MT; ++mt)
        acc[mt][nt] = __builtin_amdgcn_wmma_f32_16x16x32_bf16(
            false, a[mt].v, false, b[nt].v, (short)0, acc[mt][nt], false, false);
  }

  // C/D layout: VGPR r, lanes 0-15 -> (M=r, N=lane); lanes 16-31 -> (M=8+r, N=lane-16)
  #pragma unroll
  for (int mt = 0; mt < MT; ++mt) {
    #pragma unroll
    for (int nt = 0; nt < NT; ++nt) {
      const int col = waveN + nt * 16 + l15;
      float bs = 0.f, w3v = 0.f;
      if constexpr (EPI == 0 || EPI == 1) bs = bias[col];
      if constexpr (EPI == 1) w3v = w3[col];
      #pragma unroll
      for (int r = 0; r < 8; ++r) {
        const int m = waveM + mt * 16 + half * 8 + r;
        const float z = acc[mt][nt][r];
        if constexpr (EPI == 0) {
          const float h = tanhf(z + bs);
          Hout[(size_t)m * HSTRIDE + col] = f2bf(h);
        } else if constexpr (EPI == 1) {
          const float h = tanhf(z + bs);
          Hout[(size_t)m * HSTRIDE + col] = f2bf(w3v * (1.0f - h * h));
        } else if constexpr (EPI == 2) {
          const float hm = bf2f(Hmask[(size_t)m * HSTRIDE + col]);
          Hout[(size_t)m * HSTRIDE + col] = f2bf(z * (1.0f - hm * hm));
        } else {
          gout[(size_t)m * 16 + col] = z;
        }
      }
    }
  }
}

// ---------------------------------------------------------------------------
// Kernel 1: fused MLP forward + backward for a 128-row batch tile.
// 8 waves = 2 M-groups (64 rows) x 4 N-groups (64 cols).
// ---------------------------------------------------------------------------
__global__ __launch_bounds__(256) void mlp_fused(
    const float* __restrict__ x, const float* __restrict__ q,
    const float* __restrict__ b0, const float* __restrict__ b1,
    const float* __restrict__ b2, const float* __restrict__ W3,
    const __bf16* __restrict__ W0T, const __bf16* __restrict__ W0P,
    const __bf16* __restrict__ W1T, const __bf16* __restrict__ W2T,
    const __bf16* __restrict__ W1B, const __bf16* __restrict__ W2B,
    float* __restrict__ gradRQ, int nrows)
{
  extern __shared__ char smem_raw[];
  __bf16* A0 = (__bf16*)smem_raw;         // [128][32]   rq (padded)
  __bf16* H1 = A0 + M_TILE * 32;          // [128][264]  h1, later g1
  __bf16* H2 = H1 + M_TILE * HSTRIDE;     // [128][264]  h2, later g2
  __bf16* G3 = H2 + M_TILE * HSTRIDE;     // [128][264]  g3

  const int tid  = threadIdx.x;
  const int lane = tid & 31;
  const int wave = tid >> 5;
  const int wM   = (wave & 1) * 64;       // M-group: rows wM..wM+63
  const int wN   = (wave >> 1) * 64;      // N-group: cols wN..wN+63
  const int base = blockIdx.x * M_TILE;

  // Phase 0: build rq = [r(3), qn(8)] rows, bf16, zero-padded to K=32
  if (tid < M_TILE) {
    const int row = base + tid;
    float vals[11];
    if (row < nrows) {
      const float* xp = x + (size_t)row * 6;
      const float* qp = q + (size_t)row * 8;
      vals[0] = xp[3] - xp[0]; vals[1] = xp[4] - xp[1]; vals[2] = xp[5] - xp[2];
      #pragma unroll
      for (int p = 0; p < 2; ++p) {
        float a = qp[4*p], bq = qp[4*p+1], c = qp[4*p+2], d = qp[4*p+3];
        float inv = rsqrtf(a*a + bq*bq + c*c + d*d);
        vals[3+4*p+0]=a*inv; vals[3+4*p+1]=bq*inv; vals[3+4*p+2]=c*inv; vals[3+4*p+3]=d*inv;
      }
    } else {
      #pragma unroll
      for (int i = 0; i < 11; ++i) vals[i] = 0.0f;
    }
    __bf16* dst = A0 + tid * 32;
    #pragma unroll
    for (int i = 0; i < 11; ++i) dst[i] = f2bf(vals[i]);
    #pragma unroll
    for (int i = 11; i < 32; ++i) dst[i] = (__bf16)0.0f;
  }
  __syncthreads();

  // forward
  gemm_phase<1,4,4,0>(A0, 32,      W0T, 32,  H1, nullptr, b0, nullptr, nullptr, lane, wM, wN);
  __syncthreads();
  gemm_phase<8,4,4,0>(H1, HSTRIDE, W1T, 256, H2, nullptr, b1, nullptr, nullptr, lane, wM, wN);
  __syncthreads();
  gemm_phase<8,4,4,1>(H2, HSTRIDE, W2T, 256, G3, nullptr, b2, W3,      nullptr, lane, wM, wN);
  __syncthreads();
  // backward: g2 = (G3 @ W2^T) * (1-h2^2)  (overwrites H2), g1 likewise into H1
  gemm_phase<8,4,4,2>(G3, HSTRIDE, W2B, 256, H2, H2, nullptr, nullptr, nullptr, lane, wM, wN);
  __syncthreads();
  gemm_phase<8,4,4,2>(H2, HSTRIDE, W1B, 256, H1, H1, nullptr, nullptr, nullptr, lane, wM, wN);
  __syncthreads();
  // grad_rq = G1 @ W0^T  (N padded to 16; pad rows of W0P are zero). 16 rows/wave.
  gemm_phase<8,1,1,3>(H1, HSTRIDE, W0P, 256, nullptr, nullptr, nullptr, nullptr,
                      gradRQ + (size_t)base * 16, lane, wave * 16, 0);
}

// ---------------------------------------------------------------------------
// Kernel 2: elementwise rigid-body dynamics from grad_rq.
// Outputs concatenated: dvdt[B,2,3] | dwdt[B,2,3] | dxdt[B,2,3] | dqdt[B,2,4]
// ---------------------------------------------------------------------------
__global__ __launch_bounds__(256) void finalize(
    const float* __restrict__ v, const float* __restrict__ w,
    const float* __restrict__ x, const float* __restrict__ q,
    const float* __restrict__ gradRQ, float* __restrict__ out, int n)
{
  const int i = blockIdx.x * blockDim.x + threadIdx.x;
  if (i >= n) return;
  const float* gp = gradRQ + (size_t)i * 16;
  const float* xp = x + (size_t)i * 6;
  const float* qp = q + (size_t)i * 8;
  const float* vp = v + (size_t)i * 6;
  const float* wp = w + (size_t)i * 6;

  float r0 = xp[3]-xp[0], r1 = xp[4]-xp[1], r2 = xp[5]-xp[2];
  float rn = sqrtf(r0*r0 + r1*r1 + r2*r2);
  float coef = 100.0f * (rn - 1.0f) / rn;
  float gr0 = gp[0] + coef*r0, gr1 = gp[1] + coef*r1, gr2 = gp[2] + coef*r2;

  float* dvdt = out;
  float* dwdt = out + (size_t)n * 6;
  float* dxdt = out + (size_t)n * 12;
  float* dqdt = out + (size_t)n * 18;

  const float invM = 1.0f / 7.0f;
  dvdt[(size_t)i*6+0] =  gr0*invM; dvdt[(size_t)i*6+1] =  gr1*invM; dvdt[(size_t)i*6+2] =  gr2*invM;
  dvdt[(size_t)i*6+3] = -gr0*invM; dvdt[(size_t)i*6+4] = -gr1*invM; dvdt[(size_t)i*6+5] = -gr2*invM;
  #pragma unroll
  for (int k = 0; k < 6; ++k) dxdt[(size_t)i*6+k] = vp[k];

  const float I3[3] = {1.f, 2.f, 3.f};
  #pragma unroll
  for (int p = 0; p < 2; ++p) {
    float a = qp[4*p], b = qp[4*p+1], c = qp[4*p+2], d = qp[4*p+3];
    float inv = rsqrtf(a*a + b*b + c*c + d*d);
    float q0 = a*inv, q1 = b*inv, q2 = c*inv, q3 = d*inv;
    float wx = wp[3*p], wy = wp[3*p+1], wz = wp[3*p+2];

    // dq = 0.5 * qn * (0, w)   (Hamilton, real-first)
    float dq0 = 0.5f * (-q1*wx - q2*wy - q3*wz);
    float dq1 = 0.5f * ( q0*wx + q2*wz - q3*wy);
    float dq2 = 0.5f * ( q0*wy - q1*wz + q3*wx);
    float dq3 = 0.5f * ( q0*wz + q1*wy - q2*wx);
    dqdt[(size_t)i*8 + 4*p+0] = dq0;
    dqdt[(size_t)i*8 + 4*p+1] = dq1;
    dqdt[(size_t)i*8 + 4*p+2] = dq2;
    dqdt[(size_t)i*8 + 4*p+3] = dq3;

    float Gq[3][4] = {{-q1, q0, q3,-q2}, {-q2,-q3, q0, q1}, {-q3, q2,-q1, q0}};
    float Gd[3][4] = {{-dq1,dq0,dq3,-dq2},{-dq2,-dq3,dq0,dq1},{-dq3,dq2,-dq1,dq0}};
    float l[3]  = {wx*I3[0], wy*I3[1], wz*I3[2]};
    float gq[4] = {gp[3+4*p], gp[4+4*p], gp[5+4*p], gp[6+4*p]};
    #pragma unroll
    for (int ii = 0; ii < 3; ++ii) {
      float s = 0.f;
      #pragma unroll
      for (int kk = 0; kk < 3; ++kk) {
        float om = 2.0f * (Gq[ii][0]*Gd[kk][0] + Gq[ii][1]*Gd[kk][1] +
                           Gq[ii][2]*Gd[kk][2] + Gq[ii][3]*Gd[kk][3]);
        s += om * l[kk];
      }
      float t = Gq[ii][0]*gq[0] + Gq[ii][1]*gq[1] + Gq[ii][2]*gq[2] + Gq[ii][3]*gq[3];
      dwdt[(size_t)i*6 + 3*p + ii] = (-s - 0.5f * t) / I3[ii];
    }
  }
}

// ---------------------------------------------------------------------------
extern "C" void kernel_launch(void* const* d_in, const int* in_sizes, int n_in,
                              void* d_out, int out_size, void* d_ws, size_t ws_size,
                              hipStream_t stream)
{
  const float* v  = (const float*)d_in[0];
  const float* w  = (const float*)d_in[1];
  const float* x  = (const float*)d_in[2];
  const float* q  = (const float*)d_in[3];
  const float* W0 = (const float*)d_in[4];
  const float* b0 = (const float*)d_in[5];
  const float* W1 = (const float*)d_in[6];
  const float* b1 = (const float*)d_in[7];
  const float* W2 = (const float*)d_in[8];
  const float* b2 = (const float*)d_in[9];
  const float* W3 = (const float*)d_in[10];
  // d_in[11] = b3: constant offset on U, no effect on any output.

  const int n       = in_sizes[0] / 6;
  const int nblocks = (n + M_TILE - 1) / M_TILE;
  const int npad    = nblocks * M_TILE;

  char* ws = (char*)d_ws;
  size_t o = 0;
  __bf16* W0T = (__bf16*)(ws + o); o += (size_t)256 * 32 * 2;
  __bf16* W0P = (__bf16*)(ws + o); o += (size_t)16 * 256 * 2;
  __bf16* W1T = (__bf16*)(ws + o); o += (size_t)65536 * 2;
  __bf16* W2T = (__bf16*)(ws + o); o += (size_t)65536 * 2;
  __bf16* W1B = (__bf16*)(ws + o); o += (size_t)65536 * 2;
  __bf16* W2B = (__bf16*)(ws + o); o += (size_t)65536 * 2;
  float* gradRQ = (float*)(ws + o); o += (size_t)npad * 16 * 4;
  (void)ws_size; (void)n_in; (void)out_size;

  prep_weights<<<256, 256, 0, stream>>>(W0, W1, W2, W0T, W0P, W1T, W2T, W1B, W2B);

  const size_t lds_bytes = (size_t)(M_TILE * 32 + 3 * M_TILE * HSTRIDE) * 2; // ~206 KB, < 320 KB/WG
  (void)hipFuncSetAttribute(reinterpret_cast<const void*>(mlp_fused),
                            hipFuncAttributeMaxDynamicSharedMemorySize, (int)lds_bytes);
  mlp_fused<<<nblocks, 256, lds_bytes, stream>>>(x, q, b0, b1, b2, W3,
                                                 W0T, W0P, W1T, W2T, W1B, W2B,
                                                 gradRQ, n);

  finalize<<<(n + 255) / 256, 256, 0, stream>>>(v, w, x, q, gradRQ, (float*)d_out, n);
}
